// Model_86887188398571
// MI455X (gfx1250) — matrix operand order repaired and checked by reading, not compiled, not run on hardware
//
#include <hip/hip_runtime.h>
#include <hip/hip_bf16.h>

typedef __attribute__((ext_vector_type(16))) _Float16 v16h;
typedef __attribute__((ext_vector_type(4)))  _Float16 v4h;
typedef __attribute__((ext_vector_type(8)))  float    v8f;

#define BOARD_N 128
#define CIN     4
#define KH      64
#define COUT    24
#define KTOT    (KH * KH * CIN)            /* 16384 */
#define KITERS  (KTOT / 32)                /* 512 k-steps of 32 */
#define NTILES  2                          /* 24 couts padded to 32 = 2 x 16 */
#define MTILES  ((BOARD_N * BOARD_N) / 16) /* 1024 */
#define PADW    192                        /* padded board: rows/cols 0..190 used */
#define PADPIX  (PADW * PADW)              /* 36864 pixels */

// ---------------------------------------------------------------------------
// Kernel 1a: pack conv weights into the per-lane WMMA B-fragment layout (f16).
// B (32x16 KxN, 16-bit): lane = (Kgroup<<4)|n, halves h = K within group.
// conv_w index for k=(ky*64+kx)*4+ci, cout: k*24+cout.
// ---------------------------------------------------------------------------
__global__ void prep_wB_kernel(const float* __restrict__ conv_w,
                               _Float16* __restrict__ wsB) {
  int t = blockIdx.x * blockDim.x + threadIdx.x;     // 0 .. 2*512*32-1
  if (t >= NTILES * KITERS * 32) return;
  int lane  = t & 31;
  int iter  = (t >> 5) & (KITERS - 1);
  int ntile = t >> 14;                               // 512*32 = 2^14
  int n  = lane & 15;
  int kg = lane >> 4;
  int cout = ntile * 16 + n;
  v16h frag;
#pragma unroll
  for (int h = 0; h < 16; ++h) {
    int k = iter * 32 + ((h < 8) ? (kg * 8 + h) : (16 + kg * 8 + (h - 8)));
    float v = (cout < COUT) ? conv_w[(size_t)k * COUT + cout] : 0.0f;
    frag[h] = (_Float16)v;
  }
  *(v16h*)(wsB + (size_t)t * 16) = frag;
}

// ---------------------------------------------------------------------------
// Kernel 1b: f32 board -> f16 padded board with zero halo baked in.
// padB[(r)*192 + c] (4 halves per pixel) = board[r-31][c-31] or 0.
// SAME padding (pad_lo=31) becomes pure address arithmetic in the GEMM loop.
// ---------------------------------------------------------------------------
__global__ void prep_board_kernel(const float* __restrict__ board,
                                  _Float16* __restrict__ padB) {
  int t = blockIdx.x * blockDim.x + threadIdx.x;
  if (t >= PADPIX) return;
  int r = t / PADW;
  int c = t - r * PADW;
  int by = r - 31, bx = c - 31;
  v4h v = (v4h)(_Float16)0.0f;
  if ((unsigned)by < (unsigned)BOARD_N && (unsigned)bx < (unsigned)BOARD_N) {
    float4 f = ((const float4*)board)[(size_t)by * BOARD_N + bx];
    v[0] = (_Float16)f.x; v[1] = (_Float16)f.y;
    v[2] = (_Float16)f.z; v[3] = (_Float16)f.w;
  }
  ((v4h*)padB)[t] = v;
}

// ---------------------------------------------------------------------------
// Kernel 2: implicit-GEMM conv via V_WMMA_F32_16X16X32_F16.
// M = 16 output pixels per block, N = 16 couts per wave (2 waves/block),
// K loop = 512 WMMAs (unrolled x4). A fragment = four unguarded v4h loads
// from the padded f16 board (no cvt, no branches in the hot loop).
// A layout (16-bit 16x32): lanes 0-15: K runs {0..7,16..23}; 16-31: {8..15,24..31}
// -> per lane, two runs of 8 halves = pixels {u, u+1} and {u+4, u+5},
//    u = ky*192 + 8*(iter&7) (wave-uniform), lane base = y*192 + x + 2*g.
// ---------------------------------------------------------------------------
__global__ __launch_bounds__(64) void conv_wmma_kernel(
    const _Float16* __restrict__ padB, const float* __restrict__ conv_b,
    const _Float16* __restrict__ wsB, float* __restrict__ feat) {
  int lane  = threadIdx.x & 31;
  int ntile = threadIdx.x >> 5;          // wave 0 -> couts 0..15, wave 1 -> 16..31
  int tileM = blockIdx.x;
  int m = lane & 15, g = lane >> 4;
  int p = tileM * 16 + m;                // output pixel for this A row
  int y = p >> 7, x = p & 127;

  v8f acc = {};
  const v4h*  aBase = (const v4h*)padB + ((size_t)y * PADW + x + 2 * g);
  const v16h* bBase = (const v16h*)(wsB) + ((size_t)ntile * KITERS * 32 + lane);

  union AFrag { v16h v; v4h q[4]; };

#pragma unroll 4
  for (int iter = 0; iter < KITERS; ++iter) {
    int u = (iter >> 3) * PADW + ((iter & 7) << 3);   // wave-uniform pixel offset
    AFrag a;
    a.q[0] = aBase[u];
    a.q[1] = aBase[u + 1];
    a.q[2] = aBase[u + 4];
    a.q[3] = aBase[u + 5];
    v16h bfrag = bBase[(size_t)iter * 32];
    __builtin_prefetch(&bBase[(size_t)(iter + 4) * 32], 0, 1);
    acc = __builtin_amdgcn_wmma_f32_16x16x32_f16(
        /*neg_a=*/false, a.v, /*neg_b=*/false, bfrag,
        /*c_mod=*/(short)0, acc, /*reuse_a=*/false, /*reuse_b=*/false);
  }

  // D layout: element (m_loc = g*8 + r, n = lane&15) in acc[r]
  int n = lane & 15;
  int cout = ntile * 16 + n;
  if (cout < COUT) {
    float bias = conv_b[cout];
#pragma unroll
    for (int r = 0; r < 8; ++r) {
      int pp = tileM * 16 + g * 8 + r;
      feat[(size_t)pp * COUT + cout] = acc[r] + bias;
    }
  }
}

// ---------------------------------------------------------------------------
// Kernel 3: sequential greedy walk. One wave; weights/vals in LDS; lanes 0-7
// run one neighbor-MLP each; argmax with first-max tie rule via shfl butterfly.
// ---------------------------------------------------------------------------
__global__ __launch_bounds__(32) void walk_kernel(
    const float* __restrict__ board, const float* __restrict__ feat,
    const float* __restrict__ w1, const float* __restrict__ b1,
    const float* __restrict__ w2, const float* __restrict__ b2,
    const float* __restrict__ w3, const float* __restrict__ b3,
    const int* __restrict__ start, const int* __restrict__ end,
    float* __restrict__ out) {
  __shared__ float sW1[28 * 10], sW2[10 * 12], sW3[12 * 8];
  __shared__ float sB1[10], sB2[12], sB3[8];
  __shared__ float sVals[8][28];
  __shared__ float sAcc[28];
  __shared__ int   mdY[8], mdX[8];

  int lane = threadIdx.x;
  for (int i = lane; i < 280; i += 32) sW1[i] = w1[i];
  for (int i = lane; i < 120; i += 32) sW2[i] = w2[i];
  for (int i = lane; i < 96;  i += 32) sW3[i] = w3[i];
  if (lane < 10) sB1[lane] = b1[lane];
  if (lane < 12) sB2[lane] = b2[lane];
  if (lane < 8)  sB3[lane] = b3[lane];
  if (lane < 8) {
    // MOVES = [[1,1],[1,0],[1,-1],[0,1],[0,-1],[-1,1],[-1,0],[-1,-1]]
    mdY[lane] = (lane < 3) ? 1 : ((lane < 5) ? 0 : -1);
    mdX[lane] = (lane < 3) ? (1 - lane)
                           : ((lane == 3) ? 1 : ((lane == 4) ? -1 : (6 - lane)));
  }

  int cy = start[0], cx = start[1];
  int ey = end[0],   ex = end[1];
  if (lane < 28) {
    size_t pix = (size_t)cy * BOARD_N + cx;
    sAcc[lane] = (lane < 4) ? board[pix * CIN + lane]
                            : feat[pix * COUT + (lane - 4)];  // bias already in feat
  }
  __syncthreads();

  bool done0 = (cy == ey) && (cx == ex);
  if (!done0) {
    for (int step = 0; step < 4 * BOARD_N; ++step) {
      // gather vals[8][28] for the 8 clipped neighbors of (cy,cx)
      for (int idx = lane; idx < 8 * 28; idx += 32) {
        int n = idx / 28;
        int c = idx - n * 28;
        int gy = min(BOARD_N - 1, max(0, cy + mdY[n]));
        int gx = min(BOARD_N - 1, max(0, cx + mdX[n]));
        size_t pix = (size_t)gy * BOARD_N + gx;
        sVals[n][c] = (c < 4) ? board[pix * CIN + c]
                              : feat[pix * COUT + (c - 4)];
      }
      __syncthreads();

      float bv = -3.0e38f;
      int   bi = 0x7FFFFFFF;
      if (lane < 8) {
        float h1[10];
        for (int j = 0; j < 10; ++j) {
          float s = sB1[j];
          for (int c = 0; c < 28; ++c) s += sVals[lane][c] * sW1[c * 10 + j];
          h1[j] = fmaxf(s, 0.0f);
        }
        float h2[12];
        for (int j = 0; j < 12; ++j) {
          float s = sB2[j];
          for (int i = 0; i < 10; ++i) s += h1[i] * sW2[i * 12 + j];
          h2[j] = fmaxf(s, 0.0f);
        }
        for (int j = 0; j < 8; ++j) {
          float s = sB3[j];
          for (int i = 0; i < 12; ++i) s += h2[i] * sW3[i * 8 + j];
          s = fmaxf(s, 0.0f);
          if (s > bv) { bv = s; bi = lane * 8 + j; }  // strict > keeps first max
        }
      }
      // butterfly argmax over flat index; tie -> smaller flat index (JAX argmax)
      for (int off = 16; off > 0; off >>= 1) {
        float ov = __shfl_xor(bv, off, 32);
        int   oi = __shfl_xor(bi, off, 32);
        if (ov > bv || (ov == bv && oi < bi)) { bv = ov; bi = oi; }
      }
      int chosen = bi >> 3;

      int ny = min(BOARD_N - 1, max(0, cy + mdY[chosen]));
      int nx = min(BOARD_N - 1, max(0, cx + mdX[chosen]));
      cy = ny; cx = nx;
      if (lane < 28) sAcc[lane] += sVals[chosen][lane];
      __syncthreads();
      if (cy == ey && cx == ex) break;
    }
  }
  if (lane < 28) out[lane] = sAcc[lane];
}

// ---------------------------------------------------------------------------
// Host launcher. d_in order: board, conv_w, conv_b, w1, b1, w2, b2, w3, b3,
// start, end. d_out = s (28 f32).
// Workspace: [0,1MB) packed B frags (f16), [1MB, +288KB) padded f16 board,
// then feat (128*128*24 f32 = 1.5MB). Total ~2.9MB.
// ---------------------------------------------------------------------------
extern "C" void kernel_launch(void* const* d_in, const int* in_sizes, int n_in,
                              void* d_out, int out_size, void* d_ws, size_t ws_size,
                              hipStream_t stream) {
  const float* board  = (const float*)d_in[0];
  const float* conv_w = (const float*)d_in[1];
  const float* conv_b = (const float*)d_in[2];
  const float* w1 = (const float*)d_in[3];
  const float* b1 = (const float*)d_in[4];
  const float* w2 = (const float*)d_in[5];
  const float* b2 = (const float*)d_in[6];
  const float* w3 = (const float*)d_in[7];
  const float* b3 = (const float*)d_in[8];
  const int* start = (const int*)d_in[9];
  const int* end   = (const int*)d_in[10];

  const size_t wsB_bytes  = (size_t)NTILES * KITERS * 32 * 16 * sizeof(_Float16); // 1MB
  const size_t padB_bytes = (size_t)PADPIX * CIN * sizeof(_Float16);              // 288KB

  _Float16* wsB  = (_Float16*)d_ws;
  _Float16* padB = (_Float16*)((char*)d_ws + wsB_bytes);
  float*    feat = (float*)((char*)d_ws + wsB_bytes + padB_bytes);

  // 1a) pack weights into WMMA B-fragment layout
  prep_wB_kernel<<<(NTILES * KITERS * 32 + 255) / 256, 256, 0, stream>>>(conv_w, wsB);
  // 1b) f32 board -> padded f16 board (zero halo = SAME padding)
  prep_board_kernel<<<(PADPIX + 255) / 256, 256, 0, stream>>>(board, padB);
  // 2) implicit-GEMM conv: 1024 M-tiles x 2 waves, 512 WMMAs per wave
  conv_wmma_kernel<<<MTILES, 64, 0, stream>>>(padB, conv_b, wsB, feat);
  // 3) sequential walk, single wave
  walk_kernel<<<1, 32, 0, stream>>>(board, feat, w1, b1, w2, b2, w3, b3,
                                    start, end, (float*)d_out);
}
